// GraphAttentionLayer_18966575579324
// MI455X (gfx1250) — matrix operand order repaired and checked by reading, compiled, not run on hardware
//
#include <hip/hip_runtime.h>
#include <hip/hip_bf16.h>

#define Bn    16
#define Nn    2048
#define INC   64
#define OUTC  128
#define ALPHA 0.2f

typedef __attribute__((ext_vector_type(16))) _Float16 v16h;
typedef __attribute__((ext_vector_type(8)))  _Float16 v8h;
typedef __attribute__((ext_vector_type(8)))  float    v8f;
typedef __attribute__((ext_vector_type(4)))  float    v4f;

#define NTLOAD(p) __builtin_nontemporal_load((const v4f*)(p))

// ---------------------------------------------------------------------------
// Prep: WT16[o][k] = (f16) W[k][o]   (128 x 64, 16 KB)
// ---------------------------------------------------------------------------
__global__ __launch_bounds__(256) void k_w(const float* __restrict__ W,
                                           _Float16* __restrict__ WT16) {
  for (int idx = threadIdx.x; idx < INC * OUTC; idx += 256) {
    int k = idx & 63, o = idx >> 6;
    WT16[o * INC + k] = (_Float16)W[k * OUTC + o];
  }
}

// ---------------------------------------------------------------------------
// Kernel 1: h = features @ W via WMMA f16->f32.
// One wave per 16-row tile (4 waves / block). K = 64 -> two 16x16x32 steps.
// Outputs: h32 (f32, row-major) and hT16 (f16, [b][o][n]) -- the latter as
// one packed b128 store per o-tile straight from the C layout.
// ---------------------------------------------------------------------------
__global__ __launch_bounds__(128) void k_h(const float* __restrict__ feat,
                                           const _Float16* __restrict__ WT16,
                                           float* __restrict__ h32,
                                           _Float16* __restrict__ hT16) {
  int wave = threadIdx.x >> 5;
  int lane = threadIdx.x & 31;
  int l    = lane & 15;
  int hi   = lane >> 4;
  int kb   = hi << 3;

  int tile = blockIdx.x * 4 + wave;  // 0 .. B*N/16-1
  int b    = tile >> 7;
  int n0   = (tile & 127) << 4;
  int grow = n0 + l;

  const float* frow = feat + ((size_t)b * Nn + grow) * INC;

  v8f acc[8] = {};
#pragma unroll
  for (int kc = 0; kc < INC; kc += 32) {
    int c0 = kc + kb;
    v4f g0 = *(const v4f*)(frow + c0);
    v4f g1 = *(const v4f*)(frow + c0 + 4);
    v4f g2 = *(const v4f*)(frow + c0 + 16);
    v4f g3 = *(const v4f*)(frow + c0 + 20);
    float fv[16] = {g0.x, g0.y, g0.z, g0.w, g1.x, g1.y, g1.z, g1.w,
                    g2.x, g2.y, g2.z, g2.w, g3.x, g3.y, g3.z, g3.w};
    v16h A;
#pragma unroll
    for (int i = 0; i < 16; ++i) A[i] = (_Float16)fv[i];

    v16h Bv[8];
#pragma unroll
    for (int t = 0; t < 8; ++t)
      Bv[t] = __builtin_shufflevector(*(const v8h*)(WT16 + (16 * t + l) * INC + kc + kb), *(const v8h*)(WT16 + (16 * t + l) * INC + kc + kb + 16), 0, 1, 2, 3, 4, 5, 6, 7, 8, 9, 10, 11, 12, 13, 14, 15);
#pragma unroll
    for (int t = 0; t < 8; ++t)
      acc[t] = __builtin_amdgcn_wmma_f32_16x16x32_f16(
          false, A, false, Bv[t], (short)0, acc[t], false, false);
  }

#pragma unroll
  for (int t = 0; t < 8; ++t) {
    // h32[n][o], C layout: row n0+j+8*hi, col 16t+l
#pragma unroll
    for (int j = 0; j < 8; ++j)
      h32[((size_t)b * Nn + n0 + j + (hi << 3)) * OUTC + 16 * t + l] = acc[t][j];
    // hT16[o][n]: this lane holds 8 consecutive n for channel o=16t+l
    v8h p;
#pragma unroll
    for (int j = 0; j < 8; ++j) p[j] = (_Float16)acc[t][j];
    *(v8h*)(hT16 + ((size_t)b * OUTC + 16 * t + l) * Nn + n0 + (hi << 3)) = p;
  }
}

// ---------------------------------------------------------------------------
// Kernel 2: f1 = h @ a1, f2 = h @ a2   (one thread per (b,n), f32 accuracy)
// ---------------------------------------------------------------------------
__global__ __launch_bounds__(256) void k_f(const float* __restrict__ h32,
                                           const float* __restrict__ a,
                                           float* __restrict__ f1,
                                           float* __restrict__ f2) {
  int idx = blockIdx.x * 256 + threadIdx.x;
  if (idx >= Bn * Nn) return;
  const float* hr = h32 + (size_t)idx * OUTC;
  float s1 = 0.f, s2 = 0.f;
#pragma unroll 4
  for (int o = 0; o < OUTC; ++o) {
    float v = hr[o];
    s1 += v * a[o];
    s2 += v * a[OUTC + o];
  }
  f1[idx] = s1;
  f2[idx] = s2;
}

// ---------------------------------------------------------------------------
// Kernel 3: flash-attention masked softmax + P@h via WMMA f16->f32.
// Per-chunk issue order exploits in-order LOADcnt retirement:
//   f2(4, needed first) -> B(16, needed at WMMA) -> adj_next(4, needed next iter)
// so scores wait only for f2, the WMMA block waits only for B (whose L2
// latency is hidden under the score/exp/rescale VALU work), and the HBM adj
// prefetch keeps a full-iteration distance and is never stalled on.
// ---------------------------------------------------------------------------
__global__ __launch_bounds__(128) void k_attn(const float* __restrict__ adj,
                                              const float* __restrict__ f1,
                                              const float* __restrict__ f2,
                                              const _Float16* __restrict__ hT16,
                                              float* __restrict__ out) {
  int wave = threadIdx.x >> 5;
  int lane = threadIdx.x & 31;
  int l    = lane & 15;
  int hi   = lane >> 4;
  int kb   = hi << 3;

  int tile = blockIdx.x * 4 + wave;  // 0 .. B*N/32-1
  int b    = tile >> 7;
  int r0   = (tile & 127) << 4;
  int grow = r0 + l;

  const float* adjrow = adj + ((size_t)b * Nn + grow) * (size_t)Nn;
  const float* f2b    = f2 + b * Nn;
  float        f1v    = f1[b * Nn + grow];

  float Mrun = -1e30f;
  float Lrun = 0.0f;
  v8f   acc[8] = {};

  // prologue: chunk 0 adj in flight
  v4f aj0 = NTLOAD(adjrow + kb);
  v4f aj1 = NTLOAD(adjrow + kb + 4);
  v4f aj2 = NTLOAD(adjrow + kb + 16);
  v4f aj3 = NTLOAD(adjrow + kb + 20);

  for (int m0 = 0; m0 < Nn; m0 += 32) {
    int c0 = m0 + kb;

    // (1) f2 for this chunk (cache-hot, needed first by the scores)
    v4f q0 = *(const v4f*)(f2b + c0);
    v4f q1 = *(const v4f*)(f2b + c0 + 4);
    v4f q2 = *(const v4f*)(f2b + c0 + 16);
    v4f q3 = *(const v4f*)(f2b + c0 + 20);

    // (2) B operands: 16 x global_load_b128 from L2-resident hT16
    const _Float16* hTp = hT16 + ((size_t)b * OUTC + l) * Nn + m0 + kb;
    v16h Bv[8];
#pragma unroll
    for (int t = 0; t < 8; ++t)
      Bv[t] = __builtin_shufflevector(*(const v8h*)(hTp + (size_t)(16 * t) * Nn), *(const v8h*)(hTp + (size_t)(16 * t) * Nn + 16), 0, 1, 2, 3, 4, 5, 6, 7, 8, 9, 10, 11, 12, 13, 14, 15);

    // (3) prefetch next chunk's adj (HBM, streamed once -> non-temporal)
    int mn   = (m0 + 32 < Nn) ? (m0 + 32) : 0;  // harmless dummy on last iter
    v4f ajn0 = NTLOAD(adjrow + mn + kb);
    v4f ajn1 = NTLOAD(adjrow + mn + kb + 4);
    v4f ajn2 = NTLOAD(adjrow + mn + kb + 16);
    v4f ajn3 = NTLOAD(adjrow + mn + kb + 20);

    float av[16] = {aj0.x, aj0.y, aj0.z, aj0.w, aj1.x, aj1.y, aj1.z, aj1.w,
                    aj2.x, aj2.y, aj2.z, aj2.w, aj3.x, aj3.y, aj3.z, aj3.w};
    float qv[16] = {q0.x, q0.y, q0.z, q0.w, q1.x, q1.y, q1.z, q1.w,
                    q2.x, q2.y, q2.z, q2.w, q3.x, q3.y, q3.z, q3.w};

    float s[16];
    float tmax = -1e30f;
#pragma unroll
    for (int i = 0; i < 16; ++i) {
      int   col  = c0 + ((i < 8) ? i : (i + 8));
      float e    = f1v + qv[i];
      e          = (e > 0.f) ? e : ALPHA * e;
      bool  edge = (av[i] > 0.f) || (col == grow);  // adj + I
      s[i]       = edge ? e : -1e12f;
      tmax       = fmaxf(tmax, s[i]);
    }
    tmax       = fmaxf(tmax, __shfl_xor(tmax, 16, 32));
    float Mnew = fmaxf(Mrun, tmax);
    float r    = __expf(Mrun - Mnew);

    float psum = 0.f;
    v16h  A;
#pragma unroll
    for (int i = 0; i < 16; ++i) {
      float p = __expf(s[i] - Mnew);
      psum += p;
      A[i] = (_Float16)p;  // identity mapping onto the A VGPR layout
    }
    psum += __shfl_xor(psum, 16, 32);
    Lrun = Lrun * r + psum;
    Mrun = Mnew;

    float rb[8];
#pragma unroll
    for (int j = 0; j < 8; ++j) rb[j] = __shfl(r, j + (hi << 3), 32);
#pragma unroll
    for (int t = 0; t < 8; ++t)
#pragma unroll
      for (int j = 0; j < 8; ++j) acc[t][j] *= rb[j];

    // 8 back-to-back WMMAs, all B operands already resident
#pragma unroll
    for (int t = 0; t < 8; ++t)
      acc[t] = __builtin_amdgcn_wmma_f32_16x16x32_f16(
          false, A, false, Bv[t], (short)0, acc[t], false, false);

    aj0 = ajn0; aj1 = ajn1; aj2 = ajn2; aj3 = ajn3;
  }

  float il = 1.0f / Lrun;
  float ilb[8];
#pragma unroll
  for (int j = 0; j < 8; ++j) ilb[j] = __shfl(il, j + (hi << 3), 32);

#pragma unroll
  for (int t = 0; t < 8; ++t)
#pragma unroll
    for (int j = 0; j < 8; ++j) {
      int row = r0 + j + (hi << 3);
      out[((size_t)b * Nn + row) * OUTC + 16 * t + l] = acc[t][j] * ilb[j];
    }
}

// ---------------------------------------------------------------------------
extern "C" void kernel_launch(void* const* d_in, const int* in_sizes, int n_in,
                              void* d_out, int out_size, void* d_ws, size_t ws_size,
                              hipStream_t stream) {
  const float* feat = (const float*)d_in[0];  // (B,N,IN_C)
  const float* adj  = (const float*)d_in[1];  // (B,N,N)
  const float* W    = (const float*)d_in[2];  // (IN_C,OUT_C)
  const float* a    = (const float*)d_in[3];  // (2*OUT_C,1)
  float*       out  = (float*)d_out;          // (B,N,OUT_C)

  char*     ws   = (char*)d_ws;
  float*    h32  = (float*)ws;                                        // 16 MB
  _Float16* hT16 = (_Float16*)(ws + (size_t)Bn * Nn * OUTC * 4);      // 8 MB
  float*    f1   = (float*)(ws + (size_t)Bn * Nn * OUTC * 6);
  float*    f2   = f1 + Bn * Nn;
  _Float16* WT16 = (_Float16*)(f2 + Bn * Nn);                         // 16 KB

  k_w<<<1, 256, 0, stream>>>(W, WT16);
  k_h<<<Bn * (Nn / 64), 128, 0, stream>>>(feat, WT16, h32, hT16);
  k_f<<<(Bn * Nn + 255) / 256, 256, 0, stream>>>(h32, a, f1, f2);
  k_attn<<<Bn * (Nn / 128), 128, 0, stream>>>(adj, f1, f2, hT16, out);
}
